// GAT_7602092113942
// MI455X (gfx1250) — matrix operand order repaired and checked by reading, not compiled, not run on hardware
//
#include <hip/hip_runtime.h>
#include <hip/hip_bf16.h>
#include <math.h>

#define NODES   100000
#define EDGES   1600000
#define ETOT    (EDGES + NODES)
#define IN_F    512
#define HEADS1  8
#define CH1     8
#define H1C     64          // HEADS1 * CH1
#define OUTC    40
#define NEG_SLOPE 0.2f
#define NEGINF_KEY 0x807FFFFF   // fkey(-inf)

typedef __bf16 v16bf __attribute__((ext_vector_type(16)));
typedef float  v8f   __attribute__((ext_vector_type(8)));
typedef unsigned int u32x4 __attribute__((ext_vector_type(4)));
typedef int          i32x8 __attribute__((ext_vector_type(8)));
typedef int          i32x4 __attribute__((ext_vector_type(4)));

union FragAB { v16bf v; __bf16 e[16]; unsigned int u[8]; };

static __device__ __forceinline__ int   fkey(float f)  { int i = __float_as_int(f); return i >= 0 ? i : (i ^ 0x7fffffff); }
static __device__ __forceinline__ float funkey(int k)  { return __int_as_float(k >= 0 ? k : (k ^ 0x7fffffff)); }
static __device__ __forceinline__ float lrelu(float v) { return v > 0.0f ? v : NEG_SLOPE * v; }

static __device__ __forceinline__ void edge_get(const long long* __restrict__ ei, int e, int& s, int& d) {
  if (e < EDGES) { s = (int)ei[e]; d = (int)ei[EDGES + e]; }
  else           { s = e - EDGES; d = s; }            // self loops appended
}

// ---------------------------------------------------------------- init
__global__ void init_ws(float* agg1, float* d1, int* m1, float* out, float* d2, int* m2) {
  int i = blockIdx.x * blockDim.x + threadIdx.x;
  if (i < NODES * H1C)   agg1[i] = 0.0f;
  if (i < NODES * OUTC)  out[i]  = 0.0f;
  if (i < NODES * HEADS1){ d1[i] = 0.0f; m1[i] = NEGINF_KEY; }
  if (i < NODES)         { d2[i] = 0.0f; m2[i] = NEGINF_KEY; }
}

// ------------------------------------------------- GEMM1: h1 = x @ W1
// x-tile (16 rows x 512 f32, 32 KB) staged by the Tensor Data Mover; bf16 WMMA.
__global__ __launch_bounds__(128) void gemm1_wmma(const float* __restrict__ x,
                                                  const float* __restrict__ W1,
                                                  float* __restrict__ h1) {
  __shared__ float sxf[16 * IN_F];                     // 32 KB, filled by TDM
  const int tid = threadIdx.x;
  const int rowbase = blockIdx.x * 16;

  if (tid < 32) {                                      // wave 0 programs the TDM
    unsigned long long ga = (unsigned long long)(const void*)(x + (size_t)rowbase * IN_F);
    unsigned int lds = (unsigned int)(unsigned long long)(void*)sxf;
    u32x4 g0 = { 1u,                                   // count=1 valid user D#
                 lds,                                  // lds_addr (bytes)
                 (unsigned int)ga,                     // global_addr[31:0]
                 (unsigned int)((ga >> 32) & 0x1FFFFFFu) | (2u << 30) }; // addr[56:32] | type=2
    i32x8 g1 = { (int)(2u << 16),                      // data_size=4B, wg_mask=0
                 (int)(512u << 16),                    // tensor_dim0 = 512
                 (int)(16u << 16),                     // tensor_dim1 = 16
                 (int)(512u << 16),                    // tile_dim0 = 512
                 16,                                   // tile_dim1 = 16, tile_dim2 = 0
                 512,                                  // tensor_dim0_stride = 512
                 0, 0 };                               // tensor_dim1_stride = 0
    i32x4 g2 = { 1, 1, 0, 0 };                         // tensor_dim2=1, tensor_dim3=1
    i32x4 g3 = { 0, (int)(1u << 16), 0, 0 };           // tensor_dim4=1
    i32x8 g4 = { 0, 0, 0, 0, 0, 0, 0, 0 };             // extension words (unused)
    __builtin_amdgcn_tensor_load_to_lds(g0, g1, g2, g3, g4, 0);
    __builtin_amdgcn_s_wait_tensorcnt(0);
  }
  __syncthreads();

  const int wave = tid >> 5, lane = tid & 31;
  const int half = lane >> 4, mloc = lane & 15;
  const int ncol = wave * 16 + mloc;                   // 4 waves = 4 col tiles
  const float* arow = sxf + mloc * IN_F;

  v8f c = {};
  for (int kk = 0; kk < IN_F / 32; ++kk) {
    const int k0 = kk * 32;
    FragAB a, b;
#pragma unroll
    for (int r = 0; r < 8; ++r) {                      // A: 16-bit 16x32 layout
      const int kA = ((r & 4) << 2) + ((r & 3) << 1) + (half << 3);
      a.e[2 * r]     = (__bf16)arow[k0 + kA];
      a.e[2 * r + 1] = (__bf16)arow[k0 + kA + 1];
    }
#pragma unroll
    for (int j = 0; j < 16; ++j)                       // B: K = k0 + half*8 + j + (j&8)
      b.e[j] = (__bf16)W1[(size_t)(k0 + half * 8 + j + (j & 8)) * H1C + ncol];
    c = __builtin_amdgcn_wmma_f32_16x16x32_bf16(false, a.v, false, b.v,
                                                (short)0, c, false, false);
  }
#pragma unroll
  for (int r = 0; r < 8; ++r) {
    const int row = rowbase + half * 8 + r;            // C/D layout
    h1[(size_t)row * H1C + ncol] = c[r];
  }
}

// ------------------------------------------------- GEMM2: h2 = h @ W2 (pad 40->48)
__global__ __launch_bounds__(96) void gemm2_wmma(const float* __restrict__ h,
                                                 const float* __restrict__ W2,
                                                 float* __restrict__ h2) {
  __shared__ __bf16 sh[16 * H1C];
  const int tid = threadIdx.x;
  const int rowbase = blockIdx.x * 16;
  for (int i = tid; i < 16 * H1C; i += 96)
    sh[i] = (__bf16)h[(size_t)rowbase * H1C + i];
  __syncthreads();

  const int wave = tid >> 5, lane = tid & 31;          // wave = col tile 0..2
  const int half = lane >> 4, mloc = lane & 15;
  const int ncol = wave * 16 + mloc;
  const unsigned int* shU = (const unsigned int*)sh;

  v8f c = {};
  for (int kk = 0; kk < H1C / 32; ++kk) {
    const int k0 = kk * 32;
    FragAB a, b;
#pragma unroll
    for (int r = 0; r < 8; ++r) {
      const int kA = ((r & 4) << 2) + ((r & 3) << 1) + (half << 3);
      a.u[r] = shU[(mloc * H1C + k0 + kA) >> 1];
    }
#pragma unroll
    for (int j = 0; j < 16; ++j) {
      const int k = k0 + half * 8 + j + (j & 8);
      b.e[j] = (ncol < OUTC) ? (__bf16)W2[k * OUTC + ncol] : (__bf16)0.0f;
    }
    c = __builtin_amdgcn_wmma_f32_16x16x32_bf16(false, a.v, false, b.v,
                                                (short)0, c, false, false);
  }
  if (ncol < OUTC) {
#pragma unroll
    for (int r = 0; r < 8; ++r)
      h2[(size_t)(rowbase + half * 8 + r) * OUTC + ncol] = c[r];
  }
}

// ------------------------------------------------- attention coefficients
__global__ void att1(const float* __restrict__ h1, const float* __restrict__ a_src,
                     const float* __restrict__ a_dst, float* als, float* ald) {
  int i = blockIdx.x * blockDim.x + threadIdx.x;       // node*8 + head
  if (i >= NODES * HEADS1) return;
  int n = i >> 3, hh = i & 7;
  const float* hp = h1 + (size_t)n * H1C + hh * CH1;
  float s = 0.0f, d = 0.0f;
#pragma unroll
  for (int c = 0; c < CH1; ++c) { s += hp[c] * a_src[hh * CH1 + c]; d += hp[c] * a_dst[hh * CH1 + c]; }
  als[i] = s; ald[i] = d;
}

__global__ void att2(const float* __restrict__ h2, const float* __restrict__ a_src,
                     const float* __restrict__ a_dst, float* als, float* ald) {
  int n = blockIdx.x * blockDim.x + threadIdx.x;
  if (n >= NODES) return;
  const float* hp = h2 + (size_t)n * OUTC;
  float s = 0.0f, d = 0.0f;
  for (int c = 0; c < OUTC; ++c) { s += hp[c] * a_src[c]; d += hp[c] * a_dst[c]; }
  als[n] = s; ald[n] = d;
}

// ------------------------------------------------- layer-1 edge passes
__global__ void edge_max1(const long long* ei, const float* als, const float* ald, int* m1) {
  int e = blockIdx.x * blockDim.x + threadIdx.x;
  if (e >= ETOT) return;
  int s, d; edge_get(ei, e, s, d);
#pragma unroll
  for (int hh = 0; hh < HEADS1; ++hh)
    atomicMax(&m1[d * HEADS1 + hh], fkey(lrelu(als[s * HEADS1 + hh] + ald[d * HEADS1 + hh])));
}

__global__ void edge_sum1(const long long* ei, const float* als, const float* ald,
                          const int* m1, float* d1) {
  int e = blockIdx.x * blockDim.x + threadIdx.x;
  if (e >= ETOT) return;
  int s, d; edge_get(ei, e, s, d);
#pragma unroll
  for (int hh = 0; hh < HEADS1; ++hh) {
    float v = lrelu(als[s * HEADS1 + hh] + ald[d * HEADS1 + hh]);
    atomicAdd(&d1[d * HEADS1 + hh], __expf(v - funkey(m1[d * HEADS1 + hh])));
  }
}

__global__ void edge_agg1(const long long* ei, const float* als, const float* ald,
                          const int* m1, const float* d1,
                          const float* __restrict__ h1, float* agg) {
  int i = blockIdx.x * blockDim.x + threadIdx.x;       // edge*8 + head
  if (i >= ETOT * HEADS1) return;
  int e = i >> 3, hh = i & 7;
  int s, d; edge_get(ei, e, s, d);
  float v = lrelu(als[s * HEADS1 + hh] + ald[d * HEADS1 + hh]);
  float alpha = __expf(v - funkey(m1[d * HEADS1 + hh])) / (d1[d * HEADS1 + hh] + 1e-16f);
  const float* hp = h1 + (size_t)s * H1C + hh * CH1;
  float* op = agg + (size_t)d * H1C + hh * CH1;
#pragma unroll
  for (int c = 0; c < CH1; ++c) atomicAdd(&op[c], hp[c] * alpha);
}

__global__ void elu_bias(float* agg, const float* __restrict__ b1) {
  int i = blockIdx.x * blockDim.x + threadIdx.x;
  if (i >= NODES * H1C) return;
  float v = agg[i] + b1[i & (H1C - 1)];
  agg[i] = v > 0.0f ? v : __expf(v) - 1.0f;
}

// ------------------------------------------------- layer-2 edge passes (1 head)
__global__ void edge_max2(const long long* ei, const float* als, const float* ald, int* m2) {
  int e = blockIdx.x * blockDim.x + threadIdx.x;
  if (e >= ETOT) return;
  int s, d; edge_get(ei, e, s, d);
  atomicMax(&m2[d], fkey(lrelu(als[s] + ald[d])));
}

__global__ void edge_sum2(const long long* ei, const float* als, const float* ald,
                          const int* m2, float* d2) {
  int e = blockIdx.x * blockDim.x + threadIdx.x;
  if (e >= ETOT) return;
  int s, d; edge_get(ei, e, s, d);
  atomicAdd(&d2[d], __expf(lrelu(als[s] + ald[d]) - funkey(m2[d])));
}

__global__ void edge_agg2(const long long* ei, const float* als, const float* ald,
                          const int* m2, const float* d2,
                          const float* __restrict__ h2, float* out) {
  int i = blockIdx.x * blockDim.x + threadIdx.x;       // edge*5 + chunk (8 cols each)
  if (i >= ETOT * 5) return;
  int e = i / 5, ch = i - e * 5;
  int s, d; edge_get(ei, e, s, d);
  float v = lrelu(als[s] + ald[d]);
  float alpha = __expf(v - funkey(m2[d])) / (d2[d] + 1e-16f);
  const float* hp = h2 + (size_t)s * OUTC + ch * 8;
  float* op = out + (size_t)d * OUTC + ch * 8;
#pragma unroll
  for (int c = 0; c < 8; ++c) atomicAdd(&op[c], hp[c] * alpha);
}

// ------------------------------------------------- final: +b2, log_softmax (in place)
__global__ void logsm(float* out, const float* __restrict__ b2) {
  int n = blockIdx.x * blockDim.x + threadIdx.x;
  if (n >= NODES) return;
  float v[OUTC]; float mx = -INFINITY;
  float* row = out + (size_t)n * OUTC;
  for (int c = 0; c < OUTC; ++c) { v[c] = row[c] + b2[c]; mx = fmaxf(mx, v[c]); }
  float sum = 0.0f;
  for (int c = 0; c < OUTC; ++c) sum += __expf(v[c] - mx);
  float lse = mx + logf(sum);
  for (int c = 0; c < OUTC; ++c) row[c] = v[c] - lse;
}

static inline int cdiv(long long a, int b) { return (int)((a + b - 1) / b); }

extern "C" void kernel_launch(void* const* d_in, const int* in_sizes, int n_in,
                              void* d_out, int out_size, void* d_ws, size_t ws_size,
                              hipStream_t stream) {
  const float*     x    = (const float*)d_in[0];
  const long long* ei   = (const long long*)d_in[1];
  const float*     W1   = (const float*)d_in[2];
  const float*     a1s  = (const float*)d_in[3];
  const float*     a1d  = (const float*)d_in[4];
  const float*     b1   = (const float*)d_in[5];
  const float*     W2   = (const float*)d_in[6];
  const float*     a2s  = (const float*)d_in[7];
  const float*     a2d  = (const float*)d_in[8];
  const float*     b2   = (const float*)d_in[9];
  float* out = (float*)d_out;

  float* ws   = (float*)d_ws;
  float* h1   = ws;           ws += (size_t)NODES * H1C;
  float* als1 = ws;           ws += (size_t)NODES * HEADS1;
  float* ald1 = ws;           ws += (size_t)NODES * HEADS1;
  float* d1   = ws;           ws += (size_t)NODES * HEADS1;
  int*   m1   = (int*)ws;     ws += (size_t)NODES * HEADS1;
  float* agg1 = ws;           ws += (size_t)NODES * H1C;
  float* h2   = ws;           ws += (size_t)NODES * OUTC;
  float* als2 = ws;           ws += NODES;
  float* ald2 = ws;           ws += NODES;
  float* d2   = ws;           ws += NODES;
  int*   m2   = (int*)ws;

  init_ws<<<cdiv((long long)NODES * H1C, 256), 256, 0, stream>>>(agg1, d1, m1, out, d2, m2);

  gemm1_wmma<<<NODES / 16, 128, 0, stream>>>(x, W1, h1);
  att1<<<cdiv((long long)NODES * HEADS1, 256), 256, 0, stream>>>(h1, a1s, a1d, als1, ald1);
  edge_max1<<<cdiv(ETOT, 256), 256, 0, stream>>>(ei, als1, ald1, m1);
  edge_sum1<<<cdiv(ETOT, 256), 256, 0, stream>>>(ei, als1, ald1, m1, d1);
  edge_agg1<<<cdiv((long long)ETOT * HEADS1, 256), 256, 0, stream>>>(ei, als1, ald1, m1, d1, h1, agg1);
  elu_bias<<<cdiv((long long)NODES * H1C, 256), 256, 0, stream>>>(agg1, b1);

  gemm2_wmma<<<NODES / 16, 96, 0, stream>>>(agg1, W2, h2);
  att2<<<cdiv(NODES, 256), 256, 0, stream>>>(h2, a2s, a2d, als2, ald2);
  edge_max2<<<cdiv(ETOT, 256), 256, 0, stream>>>(ei, als2, ald2, m2);
  edge_sum2<<<cdiv(ETOT, 256), 256, 0, stream>>>(ei, als2, ald2, m2, d2);
  edge_agg2<<<cdiv((long long)ETOT * 5, 256), 256, 0, stream>>>(ei, als2, ald2, m2, d2, h2, out);

  logsm<<<cdiv(NODES, 256), 256, 0, stream>>>(out, b2);
}
